// Conceptors_47158740910797
// MI455X (gfx1250) — matrix operand order repaired and checked by reading, not compiled, not run on hardware
//
#include <hip/hip_runtime.h>

// ---------------------------------------------------------------------------
// Leaky-integrator tanh RNN on MI455X (gfx1250), persistent-kernel design.
//   S=1024 hidden, B=64 batch, O=64 outputs, L=512 steps, TAU=10.
// 16 workgroups x 256 threads (8 wave32s). WG g owns hidden columns
// [64g, 64g+64). W_r slice lives in LDS (bf16, 128KB). tanh(h) is broadcast
// per step through a double-buffered global bf16 array; a sense-reversing
// grid barrier separates steps. All matrix math via v_wmma_f32_16x16x32_bf16.
// tanh on the serial critical path uses gfx1250's hardware v_tanh_f32 when
// the builtin is available.
// ---------------------------------------------------------------------------

typedef __bf16 bf16;
typedef __attribute__((ext_vector_type(16))) __bf16 v16bf;
typedef __attribute__((ext_vector_type(8)))  __bf16 v8bf;
typedef __attribute__((ext_vector_type(8)))  float  v8f;

#define S_DIM 1024
#define B_DIM 64
#define O_DIM 64
#define L_DIM 512
#define NWG   16

#if __has_builtin(__builtin_amdgcn_tanhf)
__device__ __forceinline__ float fast_tanh(float v) { return __builtin_amdgcn_tanhf(v); }
#else
__device__ __forceinline__ float fast_tanh(float v) { return tanhf(v); }
#endif

// A-matrix (16x32 bf16, row-major source):
// lanes 0-15: K [0..15]; lanes 16-31: K [16..31] (contiguous).
__device__ __forceinline__ v16bf ldA(const bf16* p) {
  union { v16bf v; struct { v8bf lo, hi; } s; } u;
  u.s.lo = *reinterpret_cast<const v8bf*>(p);        // K = kb .. kb+7
  u.s.hi = *reinterpret_cast<const v8bf*>(p + 8);    // K = kb+8 .. kb+15
  return u.v;
}
// B-matrix (32x16 bf16): lane's 16 elems are K-contiguous down one column.
__device__ __forceinline__ v16bf ldB(const bf16* p) {
  union { v16bf v; struct { v8bf lo, hi; } s; } u;
  u.s.lo = *reinterpret_cast<const v8bf*>(p);
  u.s.hi = *reinterpret_cast<const v8bf*>(p + 8);
  return u.v;
}

__device__ __forceinline__ v8f wmma_bf16(v16bf a, v16bf b, v8f c) {
  return __builtin_amdgcn_wmma_f32_16x16x32_bf16(
      /*neg_a=*/false, a, /*neg_b=*/false, b,
      /*c_mod=*/(short)0, c, /*reuse_a=*/false, /*reuse_b=*/false);
}

__device__ __forceinline__ void grid_barrier(unsigned* cnt, unsigned* sense,
                                             unsigned nwg, unsigned phase) {
  __syncthreads();
  if (threadIdx.x == 0) {
    __threadfence();
    unsigned prev = atomicAdd(cnt, 1u);
    if (prev == nwg - 1u) {
      *cnt = 0u;
      __threadfence();
      __atomic_store_n(sense, phase, __ATOMIC_RELEASE);
    } else {
      while (__atomic_load_n(sense, __ATOMIC_ACQUIRE) < phase) {
        __builtin_amdgcn_s_sleep(1);
      }
    }
    __threadfence();
  }
  __syncthreads();
}

__global__ void rnn_init_barrier(unsigned* cnt, unsigned* sense) {
  if (threadIdx.x == 0 && blockIdx.x == 0) { *cnt = 0u; *sense = 0u; }
}

__global__ void __launch_bounds__(256)
rnn_scan(const float* __restrict__ x,       // [L,B,O]
         const float* __restrict__ h_init,  // [B,S]
         const float* __restrict__ w_r,     // [S,S]
         const float* __restrict__ b_r,     // [S]
         const float* __restrict__ w_o,     // [O,S]
         const float* __restrict__ b_o,     // [O]
         float* __restrict__ out,           // [L,B,O]
         bf16* __restrict__ th0,            // [B,S] broadcast buffer, parity 0
         bf16* __restrict__ th1,            // parity 1
         unsigned* __restrict__ bar_cnt,
         unsigned* __restrict__ bar_sense) {
  extern __shared__ char smem[];
  bf16*  wr_s = (bf16*)smem;                       // [64][1024] bf16 = 128 KB
  bf16*  wo_s = (bf16*)(smem + 64 * 1024 * 2);     // [16][1024] bf16 =  32 KB
  float* red  = (float*)(smem + 64 * 1024 * 2 + 16 * 1024 * 2); // [8][256] f32 = 8 KB

  const int g    = blockIdx.x;       // 0..15: hidden slice owner
  const int n0   = g * 64;           // first hidden column of slice
  const int tid  = threadIdx.x;
  const int wave = tid >> 5;
  const int lane = tid & 31;
  const int lr   = lane & 15;        // row/col index within a 16-wide tile
  const int lh   = lane >> 4;        // half-wave selector

  // Recurrent-GEMM tile mapping: 4 M(batch)-tiles x 4 N(hidden)-tiles over 8 waves.
  const int mt  = wave & 3;
  const int nt0 = (wave >> 2) * 2;
  const int nt1 = nt0 + 1;
  // Output-GEMM tile mapping (per WG): one 16x16 (batch, output) tile, K over waves.
  const int bt = g >> 2;             // batch tile 0..3
  const int ot = g & 3;              // output tile 0..3

  // ---- one-time: stage weight slices into LDS as bf16 --------------------
  for (int i = tid; i < 64 * S_DIM; i += 256) {
    const int n = i >> 10, k = i & 1023;
    wr_s[i] = (bf16)w_r[(n0 + n) * S_DIM + k];     // row n of slice = B column n
  }
  for (int i = tid; i < 16 * S_DIM; i += 256) {
    const int ol = i >> 10, k = i & 1023;
    wo_s[i] = (bf16)w_o[(ot * 16 + ol) * S_DIM + k];
  }
  __syncthreads();

  // ---- per-lane constants and recurrent state in registers ---------------
  const int ncol0 = n0 + nt0 * 16 + lr;
  const int ncol1 = n0 + nt1 * 16 + lr;
  const float br0 = b_r[ncol0];
  const float br1 = b_r[ncol1];

  v8f h0, h1;
  for (int j = 0; j < 8; ++j) {
    const int brow = mt * 16 + j + (lh << 3);
    h0[j] = h_init[brow * S_DIM + ncol0];
    h1[j] = h_init[brow * S_DIM + ncol1];
  }

  // preamble: publish tanh(h_init) into parity-0 buffer
  for (int j = 0; j < 8; ++j) {
    const int brow = mt * 16 + j + (lh << 3);
    th0[brow * S_DIM + ncol0] = (bf16)fast_tanh(h0[j]);
    th0[brow * S_DIM + ncol1] = (bf16)fast_tanh(h1[j]);
  }
  grid_barrier(bar_cnt, bar_sense, NWG, 1u);

  const float leak = 1.0f - 1.0f / 10.0f;
  const float itau = 1.0f / 10.0f;

  for (int t = 0; t <= L_DIM; ++t) {
    const bf16* thp = (t & 1) ? th1 : th0;   // tanh(h_t), full [B,S]
    bf16*       thn = (t & 1) ? th0 : th1;

    // ---- output phase: errors[t-1] = tanh(h_t) @ W_o^T + b_o - x[t-1] ----
    if (t >= 1) {
      v8f q = {};
      const bf16* ap = thp  + (bt * 16 + lr) * S_DIM + wave * 128 + (lh << 4);
      const bf16* bp = wo_s + lr * S_DIM + wave * 128 + (lh << 4);
#pragma unroll
      for (int ks = 0; ks < 4; ++ks)
        q = wmma_bf16(ldA(ap + ks * 32), ldB(bp + ks * 32), q);
      for (int j = 0; j < 8; ++j)
        red[wave * 256 + (j + (lh << 3)) * 16 + lr] = q[j];
      __syncthreads();
      {
        const int m = tid >> 4, n = tid & 15;
        float s = 0.0f;
#pragma unroll
        for (int w = 0; w < 8; ++w) s += red[w * 256 + tid];
        const int b  = bt * 16 + m;
        const int o  = ot * 16 + n;
        const int ix = (t - 1) * (B_DIM * O_DIM) + b * O_DIM + o;
        out[ix] = s + b_o[o] - x[ix];
      }
      __syncthreads();
    }

    if (t < L_DIM) {
      // ---- recurrent phase: acc = tanh(h_t) @ W_r_slice^T, K=1024 --------
      v8f c0 = {}, c1 = {};
      const bf16* ap  = thp  + (mt * 16 + lr) * S_DIM + (lh << 4);
      const bf16* b0p = wr_s + (nt0 * 16 + lr) * S_DIM + (lh << 4);
      const bf16* b1p = wr_s + (nt1 * 16 + lr) * S_DIM + (lh << 4);
#pragma unroll 4
      for (int ks = 0; ks < 32; ++ks) {
        const v16bf a = ldA(ap + ks * 32);
        c0 = wmma_bf16(a, ldB(b0p + ks * 32), c0);
        c1 = wmma_bf16(a, ldB(b1p + ks * 32), c1);
      }
      // h_{t+1} = leak*h_t + (1/tau)*(acc + b_r); publish tanh(h_{t+1})
      for (int j = 0; j < 8; ++j) {
        const int brow = mt * 16 + j + (lh << 3);
        h0[j] = leak * h0[j] + itau * (c0[j] + br0);
        h1[j] = leak * h1[j] + itau * (c1[j] + br1);
        thn[brow * S_DIM + ncol0] = (bf16)fast_tanh(h0[j]);
        thn[brow * S_DIM + ncol1] = (bf16)fast_tanh(h1[j]);
      }
      grid_barrier(bar_cnt, bar_sense, NWG, (unsigned)(t + 2));
    }
  }
}

extern "C" void kernel_launch(void* const* d_in, const int* in_sizes, int n_in,
                              void* d_out, int out_size, void* d_ws, size_t ws_size,
                              hipStream_t stream) {
  (void)in_sizes; (void)n_in; (void)out_size; (void)ws_size;
  const float* x      = (const float*)d_in[0];
  const float* h_init = (const float*)d_in[1];
  const float* w_r    = (const float*)d_in[2];
  const float* b_r    = (const float*)d_in[3];
  const float* w_o    = (const float*)d_in[4];
  const float* b_o    = (const float*)d_in[5];
  float* out = (float*)d_out;

  char* ws = (char*)d_ws;
  unsigned* bar_cnt   = (unsigned*)ws;
  unsigned* bar_sense = (unsigned*)(ws + 64);
  bf16* th0 = (bf16*)(ws + 1024);
  bf16* th1 = th0 + (size_t)B_DIM * S_DIM;

  rnn_init_barrier<<<1, 64, 0, stream>>>(bar_cnt, bar_sense);

  const size_t lds_bytes = 64 * 1024 * 2   // W_r slice bf16
                         + 16 * 1024 * 2   // W_o tile  bf16
                         + 8 * 256 * 4;    // cross-wave reduction f32
  (void)hipFuncSetAttribute((const void*)rnn_scan,
                            hipFuncAttributeMaxDynamicSharedMemorySize,
                            (int)lds_bytes);
  rnn_scan<<<NWG, 256, lds_bytes, stream>>>(x, h_init, w_r, b_r, w_o, b_o,
                                            out, th0, th1, bar_cnt, bar_sense);
}
